// NemotronHMOE_57647051047636
// MI455X (gfx1250) — matrix operand order repaired and checked by reading, not compiled, not run on hardware
//
#include <hip/hip_runtime.h>
#include <hip/hip_bf16.h>
#include <cstdint>

// ---------------- problem constants (match reference setup_inputs) ----------
#define E_EXPERTS 64
#define TOPK      6
#define NGRP      8
#define GSIZE     8          // E / NG
#define TKG       4
#define RSCALE    2.5f
#define CAP       512
#define DIMD      2048       // hidden D
#define IDIM      1024       // expert intermediate
#define SHDIM     4096       // shared intermediate

// ---------------- WMMA / vector types ---------------------------------------
typedef __bf16 v16bf __attribute__((ext_vector_type(16)));
typedef __bf16 v4bf  __attribute__((ext_vector_type(4)));
typedef float  v8f   __attribute__((ext_vector_type(8)));
typedef float  v4f   __attribute__((ext_vector_type(4)));
typedef int    v2i   __attribute__((ext_vector_type(2)));

#define LDT 36               // padded LDS stride (halves): 72B rows -> 8B-aligned b64
#define LDH 1026             // padded LDS stride for the 128x1024 h tile

// async global->LDS (gfx1250) when the toolchain exposes it
#if defined(__has_builtin)
#  if __has_builtin(__builtin_amdgcn_global_load_async_to_lds_b64)
#    define ASYNC_LDS 1
#  endif
#endif
#ifndef ASYNC_LDS
#  define ASYNC_LDS 0
#endif

__device__ __forceinline__ unsigned short f2bf_us(float f) {
  return __builtin_bit_cast(unsigned short, (__bf16)f);   // hw cvt
}
__device__ __forceinline__ __bf16 us2bf(unsigned short u) {
  return __builtin_bit_cast(__bf16, u);
}
// packed fp32x4 -> bf16x4 (2x v_cvt_pk_bf16_f32, result is one b64)
__device__ __forceinline__ unsigned long long cvt4(v4f f) {
  const v4bf b = __builtin_convertvector(f, v4bf);
  return __builtin_bit_cast(unsigned long long, b);
}
__device__ __forceinline__ v8f wmma_bf16(v16bf a, v16bf b, v8f c) {
  return __builtin_amdgcn_wmma_f32_16x16x32_bf16(
      false, a, false, b, (short)0, c, false, false);
}
__device__ __forceinline__ void wait_async0() {
#if defined(__has_builtin) && __has_builtin(__builtin_amdgcn_s_wait_asynccnt)
  __builtin_amdgcn_s_wait_asynccnt(0);
#else
  asm volatile("s_wait_asynccnt 0x0" ::: "memory");
#endif
}
#if ASYNC_LDS
__device__ __forceinline__ void async_cp_b64(const void* g, void* l) {
  // builtin signature (probe-diagnosed): (v2i AS1* gsrc, v2i AS3* ldst, i32 off, i32 cpol)
  auto gp = (__attribute__((address_space(1))) v2i*)(unsigned long long)(uintptr_t)g;
  auto lp = (__attribute__((address_space(3))) v2i*)(unsigned int)(uintptr_t)l;  // flat LDS: low 32b = offset
  __builtin_amdgcn_global_load_async_to_lds_b64(gp, lp, 0, 0);
}
#endif

// A fragment: 16x32 bf16 rows row0.., K window [k0,k0+32) from LDS (stride lda)
// ISA 7.12.2: lane m = L%16; VGPR j: K = (j>=4?16:0) + (L>=16?8:0) + (j&3)*2 + p
__device__ __forceinline__ v16bf load_frag_a(const unsigned short* lds,
                                             int row0, int k0, int lda, int lane) {
  v16bf a;
  const int m   = row0 + (lane & 15);
  const int khi = (lane >= 16) ? 8 : 0;
#pragma unroll
  for (int h = 0; h < 16; ++h) {
    const int j = h >> 1, p = h & 1;
    const int k = ((j >= 4) ? 16 : 0) + khi + ((j & 3) << 1) + p;
    a[h] = us2bf(lds[m * lda + k0 + k]);
  }
  return a;
}
// B fragment: 32x16 bf16; weight tile stored [n][k] row-major in LDS.
__device__ __forceinline__ v16bf load_frag_b(const unsigned short* lds,
                                             int n0, int ldb, int lane) {
  v16bf b;
  const int n   = n0 + (lane & 15);
  const int khi = (lane >= 16) ? 8 : 0;
#pragma unroll
  for (int h = 0; h < 16; ++h)
    b[h] = us2bf(lds[n * ldb + khi + h + ((h >= 8) ? 8 : 0)]);
  return b;
}

// ---- slab staging: 128 rows x 32 K, 256 threads, 4 x (float4 | b64) each ---
template <bool NT>
__device__ __forceinline__ void slab_load_f32(const float* __restrict__ base, size_t ld,
                                              int kb, int tid, v4f v[4]) {
#pragma unroll
  for (int i = 0; i < 4; ++i) {
    const int idx = tid + i * 256;
    const int r = idx >> 3, c4 = idx & 7;
    const v4f* p = (const v4f*)&base[(size_t)r * ld + kb + c4 * 4];
    if (NT) v[i] = __builtin_nontemporal_load(p);   // stream weights past L2 residents
    else    v[i] = *p;
  }
}
__device__ __forceinline__ void slab_load_gather(const float* __restrict__ x,
                                                 const int* __restrict__ sTok,
                                                 int kb, int tid, v4f v[4]) {
#pragma unroll
  for (int i = 0; i < 4; ++i) {
    const int idx = tid + i * 256;
    const int r = idx >> 3, c4 = idx & 7;
    const int tk = sTok[r];
    v[i] = (tk >= 0) ? *(const v4f*)&x[(size_t)tk * DIMD + kb + c4 * 4]
                     : v4f{0.f, 0.f, 0.f, 0.f};
  }
}
__device__ __forceinline__ void slab_store_bf16(unsigned short* lds, int tid, const v4f v[4]) {
#pragma unroll
  for (int i = 0; i < 4; ++i) {
    const int idx = tid + i * 256;
    const int r = idx >> 3, c4 = idx & 7;
    *(unsigned long long*)&lds[r * LDT + c4 * 4] = cvt4(v[i]);
  }
}
__device__ __forceinline__ void slab_load_raw(const unsigned short* __restrict__ base, size_t ld,
                                              int kb, int tid, unsigned long long v[4]) {
#pragma unroll
  for (int i = 0; i < 4; ++i) {
    const int idx = tid + i * 256;
    const int r = idx >> 3, c4 = idx & 7;
    v[i] = *(const unsigned long long*)&base[(size_t)r * ld + kb + c4 * 4];
  }
}
__device__ __forceinline__ void slab_store_raw(unsigned short* lds, int tid,
                                               const unsigned long long v[4]) {
#pragma unroll
  for (int i = 0; i < 4; ++i) {
    const int idx = tid + i * 256;
    const int r = idx >> 3, c4 = idx & 7;
    *(unsigned long long*)&lds[r * LDT + c4 * 4] = v[i];
  }
}
#if ASYNC_LDS
// pure bf16 copy global->LDS slab, no VGPR round trip (ASYNCcnt tracked)
__device__ __forceinline__ void slab_async_raw(const unsigned short* __restrict__ base, size_t ld,
                                               int kb, int tid, unsigned short* lds) {
#pragma unroll
  for (int i = 0; i < 4; ++i) {
    const int idx = tid + i * 256;
    const int r = idx >> 3, c4 = idx & 7;
    async_cp_b64(&base[(size_t)r * ld + kb + c4 * 4], &lds[r * LDT + c4 * 4]);
  }
}
#endif

// ---- one 32-K MMA step: wave (wr,wc) owns a 32x64 tile of the 128x128 block
__device__ __forceinline__ void mma_step(const unsigned short* bufA, int ldA, int k0A,
                                         const unsigned short* bufB,
                                         int wr, int wc, int lane, v8f acc[2][4]) {
  const v16bf a0 = load_frag_a(bufA, wr * 32,      k0A, ldA, lane);
  const v16bf a1 = load_frag_a(bufA, wr * 32 + 16, k0A, ldA, lane);
#pragma unroll
  for (int ct = 0; ct < 4; ++ct) {
    const v16bf b = load_frag_b(bufB, wc * 64 + ct * 16, LDT, lane);
    acc[0][ct] = wmma_bf16(a0, b, acc[0][ct]);
    acc[1][ct] = wmma_bf16(a1, b, acc[1][ct]);
  }
}

// ---------------- kernel 0: zero dispatch counters --------------------------
__global__ void k_zero(int* __restrict__ cnt) {
  if (threadIdx.x < E_EXPERTS) cnt[threadIdx.x] = 0;
}

// ---------------- kernel 1: router + dispatch -------------------------------
__global__ void k_router(const float* __restrict__ x, const float* __restrict__ gw,
                         const float* __restrict__ ebias,
                         int* __restrict__ cnt, int* __restrict__ tokl,
                         float* __restrict__ gatel, int T) {
  const int tok = blockIdx.x * blockDim.x + threadIdx.x;
  if (tok >= T) return;
  float raw[E_EXPERTS], sc[E_EXPERTS];
  const float* xr = x + (size_t)tok * DIMD;
#pragma unroll 1
  for (int e = 0; e < E_EXPERTS; ++e) {
    const float* wr = gw + (size_t)e * DIMD;
    float acc = 0.f;
    for (int k = 0; k < DIMD; k += 4)
      acc += xr[k] * wr[k] + xr[k + 1] * wr[k + 1] +
             xr[k + 2] * wr[k + 2] + xr[k + 3] * wr[k + 3];
    const float s = 1.f / (1.f + __expf(-acc));   // sigmoid
    raw[e] = s;
    sc[e]  = s + ebias[e];
  }
  float grp[NGRP];
#pragma unroll
  for (int g = 0; g < NGRP; ++g) {                // sum of top-2 per group
    float m1 = -1e30f, m2 = -1e30f;
#pragma unroll
    for (int i = 0; i < GSIZE; ++i) {
      const float v = sc[g * GSIZE + i];
      if (v > m1) { m2 = m1; m1 = v; } else if (v > m2) { m2 = v; }
    }
    grp[g] = m1 + m2;
  }
  unsigned gmask = 0;
  for (int t = 0; t < TKG; ++t) {                 // top-4 groups
    int best = 0; float bv = -1e30f;
#pragma unroll
    for (int g = 0; g < NGRP; ++g)
      if (!((gmask >> g) & 1u) && grp[g] > bv) { bv = grp[g]; best = g; }
    gmask |= 1u << best;
  }
  int eidx[TOPK]; float wts[TOPK]; float wsum = 0.f;
  unsigned long long used = 0ull;
  for (int t = 0; t < TOPK; ++t) {                // top-6 experts, raw-score weights
    int be = 0; float bv = -1e30f;
#pragma unroll 1
    for (int e = 0; e < E_EXPERTS; ++e) {
      if (!((gmask >> (e / GSIZE)) & 1u)) continue;
      if ((used >> e) & 1ull) continue;
      if (sc[e] > bv) { bv = sc[e]; be = e; }
    }
    used |= 1ull << be;
    eidx[t] = be; wts[t] = raw[be]; wsum += raw[be];
  }
  const float inv = RSCALE / (wsum + 1e-20f);
#pragma unroll
  for (int t = 0; t < TOPK; ++t) {
    const int e = eidx[t];
    const int slot = atomicAdd(&cnt[e], 1);
    if (slot < CAP) {
      tokl[e * CAP + slot]  = tok;
      gatel[e * CAP + slot] = wts[t] * inv;
    }
  }
}

// ---------------- kernel 2: shared up-proj, relu^2, bf16 out ----------------
__global__ void __launch_bounds__(256)
k_shared_up(const float* __restrict__ x, const float* __restrict__ su,
            unsigned short* __restrict__ hs) {
  __shared__ unsigned short sA[2 * 128 * LDT];
  __shared__ unsigned short sB[2 * 128 * LDT];
  const int tid = threadIdx.x, lane = tid & 31, wave = tid >> 5;
  const int wr = wave >> 1, wc = wave & 1;
  const int row0 = blockIdx.x * 128, col0 = blockIdx.y * 128;
  const float* Ab = x  + (size_t)row0 * DIMD;
  const float* Bb = su + (size_t)col0 * DIMD;
  v8f acc[2][4] = {};
  {
    v4f va[4], vb[4];
    slab_load_f32<false>(Ab, DIMD, 0, tid, va);
    slab_load_f32<false>(Bb, DIMD, 0, tid, vb);
    slab_store_bf16(sA, tid, va);
    slab_store_bf16(sB, tid, vb);
  }
  __syncthreads();
  for (int kb = 0; kb < DIMD; kb += 32) {
    const int cur = (kb >> 5) & 1, nxt = cur ^ 1;
    const bool more = (kb + 32) < DIMD;
    v4f va[4], vb[4];
    if (more) { slab_load_f32<false>(Ab, DIMD, kb + 32, tid, va);
                slab_load_f32<false>(Bb, DIMD, kb + 32, tid, vb); }
    mma_step(sA + cur * 128 * LDT, LDT, 0, sB + cur * 128 * LDT, wr, wc, lane, acc);
    if (more) { slab_store_bf16(sA + nxt * 128 * LDT, tid, va);
                slab_store_bf16(sB + nxt * 128 * LDT, tid, vb); }
    __syncthreads();
  }
  const int mrow = (lane >= 16) ? 8 : 0;
  const int nn = lane & 15;
#pragma unroll
  for (int ar = 0; ar < 2; ++ar)
#pragma unroll
    for (int ct = 0; ct < 4; ++ct)
#pragma unroll
      for (int r = 0; r < 8; ++r) {
        float v = acc[ar][ct][r];
        v = (v > 0.f) ? v * v : 0.f;              // relu^2
        const int m = row0 + wr * 32 + ar * 16 + mrow + r;
        const int n = col0 + wc * 64 + ct * 16 + nn;
        hs[(size_t)m * SHDIM + n] = f2bf_us(v);
      }
}

// ---------------- kernel 3: shared down-proj, writes d_out ------------------
__global__ void __launch_bounds__(256)
k_shared_down(const unsigned short* __restrict__ hs, const float* __restrict__ sd,
              float* __restrict__ out) {
  __shared__ unsigned short sA[2 * 128 * LDT];
  __shared__ unsigned short sB[2 * 128 * LDT];
  const int tid = threadIdx.x, lane = tid & 31, wave = tid >> 5;
  const int wr = wave >> 1, wc = wave & 1;
  const int row0 = blockIdx.x * 128, col0 = blockIdx.y * 128;
  const unsigned short* Ab = hs + (size_t)row0 * SHDIM;   // already bf16
  const float* Bb = sd + (size_t)col0 * SHDIM;
  v8f acc[2][4] = {};
  {
    v4f vb[4];
    slab_load_f32<false>(Bb, SHDIM, 0, tid, vb);
#if ASYNC_LDS
    slab_async_raw(Ab, SHDIM, 0, tid, sA);                // DMA bf16 -> LDS
#else
    unsigned long long va[4];
    slab_load_raw(Ab, SHDIM, 0, tid, va);
    slab_store_raw(sA, tid, va);
#endif
    slab_store_bf16(sB, tid, vb);
#if ASYNC_LDS
    wait_async0();
#endif
  }
  __syncthreads();
  for (int kb = 0; kb < SHDIM; kb += 32) {
    const int cur = (kb >> 5) & 1, nxt = cur ^ 1;
    const bool more = (kb + 32) < SHDIM;
    v4f vb[4];
#if ASYNC_LDS
    if (more) slab_async_raw(Ab, SHDIM, kb + 32, tid, sA + nxt * 128 * LDT);
#else
    unsigned long long va[4];
    if (more) slab_load_raw(Ab, SHDIM, kb + 32, tid, va);
#endif
    if (more) slab_load_f32<false>(Bb, SHDIM, kb + 32, tid, vb);
    mma_step(sA + cur * 128 * LDT, LDT, 0, sB + cur * 128 * LDT, wr, wc, lane, acc);
    if (more) {
#if !ASYNC_LDS
      slab_store_raw(sA + nxt * 128 * LDT, tid, va);
#endif
      slab_store_bf16(sB + nxt * 128 * LDT, tid, vb);
#if ASYNC_LDS
      wait_async0();                                      // publish DMA before barrier
#endif
    }
    __syncthreads();
  }
  const int mrow = (lane >= 16) ? 8 : 0;
  const int nn = lane & 15;
#pragma unroll
  for (int ar = 0; ar < 2; ++ar)
#pragma unroll
    for (int ct = 0; ct < 4; ++ct)
#pragma unroll
      for (int r = 0; r < 8; ++r) {
        const int m = row0 + wr * 32 + ar * 16 + mrow + r;
        const int n = col0 + wc * 64 + ct * 16 + nn;
        out[(size_t)m * DIMD + n] = acc[ar][ct][r];
      }
}

// ---------------- kernel 4: fused routed expert (up -> relu^2 -> down) ------
// 128x1024 h tile stays in LDS as bf16 (~257 KB of the 320 KB WGP LDS).
__global__ void __launch_bounds__(256)
k_expert(const float* __restrict__ x, const float* __restrict__ w1,
         const float* __restrict__ w2, const int* __restrict__ cnt,
         const int* __restrict__ tokl, const float* __restrict__ gatel,
         float* __restrict__ out) {
  extern __shared__ unsigned char smem[];
  unsigned short* sH    = (unsigned short*)smem;           // 128 x LDH
  unsigned short* sA    = sH + 128 * LDH;                  // 2 x 128 x LDT
  unsigned short* sB    = sA + 2 * 128 * LDT;              // 2 x 128 x LDT
  int*            sTok  = (int*)(sB + 2 * 128 * LDT);      // 128
  float*          sGate = (float*)(sTok + 128);            // 128

  const int e  = blockIdx.x;
  const int m0 = blockIdx.y * 128;
  int n_e = cnt[e]; if (n_e > CAP) n_e = CAP;
  if (m0 >= n_e) return;
  const int rows = min(128, n_e - m0);

  const int tid = threadIdx.x, lane = tid & 31, wave = tid >> 5;
  const int wr = wave >> 1, wc = wave & 1;
  if (tid < 128) {
    if (tid < rows) { sTok[tid] = tokl[e * CAP + m0 + tid]; sGate[tid] = gatel[e * CAP + m0 + tid]; }
    else            { sTok[tid] = -1;                        sGate[tid] = 0.f; }
  }
  __syncthreads();

  const float* W1 = w1 + (size_t)e * IDIM * DIMD;   // [I, D]
  const float* W2 = w2 + (size_t)e * DIMD * IDIM;   // [D, I]

  // ---- up: h = relu2( gather(X) @ W1^T ), kept in LDS as bf16 ----
  for (int nb = 0; nb < IDIM; nb += 128) {
    const float* Bb = W1 + (size_t)nb * DIMD;
    v8f acc[2][4] = {};
    {
      v4f va[4], vb[4];
      slab_load_gather(x, sTok, 0, tid, va);
      slab_load_f32<true>(Bb, DIMD, 0, tid, vb);
      slab_store_bf16(sA, tid, va);
      slab_store_bf16(sB, tid, vb);
    }
    __syncthreads();
    for (int kb = 0; kb < DIMD; kb += 32) {
      const int cur = (kb >> 5) & 1, nxt = cur ^ 1;
      const bool more = (kb + 32) < DIMD;
      v4f va[4], vb[4];
      if (more) { slab_load_gather(x, sTok, kb + 32, tid, va);
                  slab_load_f32<true>(Bb, DIMD, kb + 32, tid, vb); }   // NT weight stream
      mma_step(sA + cur * 128 * LDT, LDT, 0, sB + cur * 128 * LDT, wr, wc, lane, acc);
      if (more) { slab_store_bf16(sA + nxt * 128 * LDT, tid, va);
                  slab_store_bf16(sB + nxt * 128 * LDT, tid, vb); }
      __syncthreads();
    }
    const int mrow = (lane >= 16) ? 8 : 0;
    const int nn = lane & 15;
#pragma unroll
    for (int ar = 0; ar < 2; ++ar)
#pragma unroll
      for (int ct = 0; ct < 4; ++ct)
#pragma unroll
        for (int r = 0; r < 8; ++r) {
          float v = acc[ar][ct][r];
          v = (v > 0.f) ? v * v : 0.f;             // relu^2
          sH[(wr * 32 + ar * 16 + mrow + r) * LDH + nb + wc * 64 + ct * 16 + nn] = f2bf_us(v);
        }
    __syncthreads();
  }

  // ---- down: y = h @ W2^T ; gate-scaled atomic accumulate into out ----
  for (int nb = 0; nb < DIMD; nb += 128) {
    const float* Bb = W2 + (size_t)nb * IDIM;
    v8f acc[2][4] = {};
    {
      v4f vb[4];
      slab_load_f32<true>(Bb, IDIM, 0, tid, vb);
      slab_store_bf16(sB, tid, vb);
    }
    __syncthreads();
    for (int kb = 0; kb < IDIM; kb += 32) {
      const int cur = (kb >> 5) & 1, nxt = cur ^ 1;
      const bool more = (kb + 32) < IDIM;
      v4f vb[4];
      if (more) slab_load_f32<true>(Bb, IDIM, kb + 32, tid, vb);       // NT weight stream
      mma_step(sH, LDH, kb, sB + cur * 128 * LDT, wr, wc, lane, acc);  // A straight from h
      if (more) slab_store_bf16(sB + nxt * 128 * LDT, tid, vb);
      __syncthreads();
    }
    const int mrow = (lane >= 16) ? 8 : 0;
    const int nn = lane & 15;
#pragma unroll
    for (int ar = 0; ar < 2; ++ar)
#pragma unroll
      for (int ct = 0; ct < 4; ++ct)
#pragma unroll
        for (int r = 0; r < 8; ++r) {
          const int m = wr * 32 + ar * 16 + mrow + r;
          if (m < rows) {
            const float v = acc[ar][ct][r] * sGate[m];
            atomicAdd(&out[(size_t)sTok[m] * DIMD + nb + wc * 64 + ct * 16 + nn], v);
          }
        }
  }
}

// ---------------- host launcher --------------------------------------------
extern "C" void kernel_launch(void* const* d_in, const int* in_sizes, int n_in,
                              void* d_out, int out_size, void* d_ws, size_t ws_size,
                              hipStream_t stream) {
  const float* x  = (const float*)d_in[0];
  const float* gw = (const float*)d_in[1];
  const float* eb = (const float*)d_in[2];
  const float* w1 = (const float*)d_in[3];
  const float* w2 = (const float*)d_in[4];
  const float* su = (const float*)d_in[5];
  const float* sd = (const float*)d_in[6];
  float* out = (float*)d_out;
  const int T = in_sizes[0] / DIMD;

  unsigned char* ws = (unsigned char*)d_ws;
  int*   cnt   = (int*)ws;                                              // E
  int*   tokl  = (int*)(ws + 256);                                      // E*CAP
  float* gatel = (float*)(ws + 256 + (size_t)E_EXPERTS * CAP * 4);      // E*CAP
  unsigned short* hs = (unsigned short*)(ws + 256 + (size_t)E_EXPERTS * CAP * 8); // T*SH bf16

  k_zero<<<1, 64, 0, stream>>>(cnt);
  k_router<<<(T + 127) / 128, 128, 0, stream>>>(x, gw, eb, cnt, tokl, gatel, T);

  dim3 gU(T / 128, SHDIM / 128);
  k_shared_up<<<gU, 256, 0, stream>>>(x, su, hs);

  dim3 gD(T / 128, DIMD / 128);
  k_shared_down<<<gD, 256, 0, stream>>>(hs, sd, out);     // initializes d_out

  const int smem = (128 * LDH + 4 * 128 * LDT) * 2 + 128 * 4 + 128 * 4;  // ~300.5 KB
  (void)hipFuncSetAttribute((const void*)k_expert,
                            hipFuncAttributeMaxDynamicSharedMemorySize, smem);
  dim3 gE(E_EXPERTS, CAP / 128);
  k_expert<<<gE, 256, (size_t)smem, stream>>>(x, w1, w2, cnt, tokl, gatel, out);
}
